// NNK_Asym_Kernel_17274358465085
// MI455X (gfx1250) — compile-verified, matrix operand order checked by reading, not run on hardware
//
#include <hip/hip_runtime.h>
#include <hip/hip_bf16.h>
#include <hip/hip_fp16.h>

// ---------------------------------------------------------------------------
// Problem constants (from reference)
// ---------------------------------------------------------------------------
#define DIMX  256      // input feature dim
#define RFS   256      // num random features
#define OUTD  1024     // output dim
#define NROW  65536    // batch rows
#define KBIG  1024     // concat dim of real-ified feature space (2R real + 2R imag)

// sqrt(1+4M) with M=0.001
#define SCALEF  1.0019980049f
// c / sqrt(R) = (1/sqrt(2))*(1.004)^(256/4) / 16
#define SFEAT   0.057058636f

typedef _Float16 v16h __attribute__((ext_vector_type(16)));
typedef _Float16 v8h  __attribute__((ext_vector_type(8)));
typedef float    v8f  __attribute__((ext_vector_type(8)));

// ---------------------------------------------------------------------------
// WMMA helpers (gfx1250, wave32)
// ---------------------------------------------------------------------------
__device__ __forceinline__ v8f wmma16(v16h a, v16h b, v8f c) {
    // emits v_wmma_f32_16x16x32_f16
    return __builtin_amdgcn_wmma_f32_16x16x32_f16(
        /*neg_a=*/false, a, /*neg_b=*/false, b,
        /*c_mod=*/(short)0, c, /*reuse_a=*/false, /*reuse_b=*/false);
}

// Load a 16x32 (f16) fragment from a row-major LDS tile.
// Per ISA 16-bit A layout: lanes 0-15 rows 0-15 / K {0..7,16..23},
// lanes 16-31 rows 0-15 / K {8..15,24..31}.  Two contiguous 16B ds loads.
__device__ __forceinline__ v16h ldsFrag(const _Float16* base, int strideH, int lane) {
    int r  = lane & 15;
    int kb = (lane >> 4) << 3;
    const _Float16* p = base + r * strideH;
    v16h a;
    *reinterpret_cast<v8h*>(&a)       = *reinterpret_cast<const v8h*>(p + kb);
    *(reinterpret_cast<v8h*>(&a) + 1) = *reinterpret_cast<const v8h*>(p + 16 + kb);
    return a;
}

// ---------------------------------------------------------------------------
// CDNA5 async global->LDS copy (VFLAT, tracked with ASYNCcnt).
// Each active lane moves 16 bytes from its global address to its LDS address;
// no VGPR data registers, no scratch spills.
// ---------------------------------------------------------------------------
__device__ __forceinline__ unsigned ldsAddr(const void* p) {
    // Truncate generic (flat) shared pointer to its 32-bit LDS offset.
    return (unsigned)(unsigned long long)(uintptr_t)p;
}
__device__ __forceinline__ void asyncCopy16(unsigned lds, const void* g) {
    unsigned long long ga = (unsigned long long)(uintptr_t)g;
    asm volatile("global_load_async_to_lds_b128 %0, %1, off"
                 :: "v"(lds), "v"(ga) : "memory");
}
__device__ __forceinline__ void waitAsync0() {
#if __has_builtin(__builtin_amdgcn_s_wait_asynccnt)
    __builtin_amdgcn_s_wait_asynccnt(0);
#else
    asm volatile("s_wait_asynccnt 0x0" ::: "memory");
#endif
}

// ---------------------------------------------------------------------------
// Kernel 1: per-RF constants + proj -> fp16
//   rfc[r*8+ {0..6}] = { scale*u, scale*v, (u^2-v^2)/2, u*v, M*||g_r||^2, u, v }
// ---------------------------------------------------------------------------
__global__ __launch_bounds__(256) void rf_kernel(const float* __restrict__ proj,
                                                 const float* __restrict__ xr,
                                                 const float* __restrict__ xi,
                                                 float* __restrict__ rfc,
                                                 _Float16* __restrict__ Ph) {
    int r = blockIdx.x, d = threadIdx.x;
    float p = proj[r * DIMX + d];
    Ph[r * DIMX + d] = (_Float16)p;
    __shared__ float red[256];
    red[d] = p * p;
    __syncthreads();
    for (int s = 128; s > 0; s >>= 1) {
        if (d < s) red[d] += red[d + s];
        __syncthreads();
    }
    if (d == 0) {
        float g2 = red[0];
        float u = xr[r], v = xi[r];
        float* c = rfc + r * 8;
        c[0] = SCALEF * u;
        c[1] = SCALEF * v;
        c[2] = 0.5f * (u * u - v * v);
        c[3] = u * v;
        c[4] = 0.001f * g2;
        c[5] = u;
        c[6] = v;
    }
}

// ---------------------------------------------------------------------------
// Kernel 2: x -> fp16 (scaled by 1/256) + per-row squared norm (wave32 reduce)
// ---------------------------------------------------------------------------
__global__ __launch_bounds__(256) void xprep_kernel(const float* __restrict__ x,
                                                    _Float16* __restrict__ Xh,
                                                    float* __restrict__ sqx) {
    int lane = threadIdx.x & 31;
    int n    = blockIdx.x * 8 + (threadIdx.x >> 5);
    const float* row  = x  + (size_t)n * DIMX;
    _Float16*    orow = Xh + (size_t)n * DIMX;
    float s = 0.f;
#pragma unroll
    for (int i = 0; i < 8; ++i) {
        float v = row[lane + i * 32] * (1.0f / 256.0f);
        orow[lane + i * 32] = (_Float16)v;
        s += v * v;
    }
#pragma unroll
    for (int off = 16; off > 0; off >>= 1) s += __shfl_xor(s, off, 32);
    if (lane == 0) sqx[n] = s;
}

// ---------------------------------------------------------------------------
// Kernel 3: weight-side features -> W (OUTD x KBIG, fp16)
//   k layout: [0..255]=Re w+, [256..511]=Re w-, [512..767]=-Im w+, [768..1023]=-Im w-
// ---------------------------------------------------------------------------
__global__ __launch_bounds__(256) void wfeat_kernel(const float* __restrict__ iw,
                                                    const float* __restrict__ bvec,
                                                    const float* __restrict__ proj,
                                                    const float* __restrict__ rfc,
                                                    _Float16* __restrict__ WB) {
    int o = blockIdx.x, tid = threadIdx.x;
    __shared__ float wrow[256];
    __shared__ float red[256];
    float wv = iw[o * DIMX + tid] * 0.25f;   // inv4 = 256^-0.25 = 1/4
    wrow[tid] = wv;
    red[tid]  = wv * wv;
    __syncthreads();
    for (int s = 128; s > 0; s >>= 1) {
        if (tid < s) red[tid] += red[tid + s];
        __syncthreads();
    }
    float sq = red[0];
    float bb = bvec[o] * 0.25f;
    int r = tid;
    float dot = 0.f;
#pragma unroll 8
    for (int d = 0; d < DIMX; ++d) dot += wrow[d] * proj[r * DIMX + d];
    const float* c = rfc + r * 8;
    float cdr = c[0], cdi = c[1], hr = c[2], hi = c[3], mg2 = c[4], u = c[5], v = c[6];
    float are =  cdr * dot - sq * hr - mg2 + u * bb;
    float aim =  cdi * dot - sq * hi + v * bb;
    float bre = -cdr * dot - sq * hr - mg2 - u * bb;
    float bim = -cdi * dot - sq * hi - v * bb;
    float ep = SFEAT * __expf(are);
    float em = SFEAT * __expf(bre);
    float sp, cp, sm, cm;
    __sincosf(aim, &sp, &cp);
    __sincosf(bim, &sm, &cm);
    _Float16* wr = WB + (size_t)o * KBIG;
    wr[r]        = (_Float16)(ep * cp);
    wr[256 + r]  = (_Float16)(em * cm);
    wr[512 + r]  = (_Float16)(-(ep * sp));
    wr[768 + r]  = (_Float16)(-(em * sm));
}

// ---------------------------------------------------------------------------
// Kernel 4: fused  P = Xh @ Ph^T  (WMMA, K=256)  ->  exp/sincos  ->  F (fp16)
//   Block: 64 x-rows, all 256 RFs. 8 waves as 2(M) x 4(N); wave tile 32x64.
//   Staging via async global->LDS copies, double buffered.
// ---------------------------------------------------------------------------
__global__ __launch_bounds__(256) void xfeat_kernel(const _Float16* __restrict__ Xh,
                                                    const _Float16* __restrict__ Ph,
                                                    const float* __restrict__ rfc,
                                                    const float* __restrict__ sqx,
                                                    _Float16* __restrict__ F) {
    __shared__ __align__(16) _Float16 As[2][64][32];    //  8 KB
    __shared__ __align__(16) _Float16 Bs[2][256][32];   // 32 KB
    int tid = threadIdx.x, lane = tid & 31, wave = tid >> 5;
    int wm = wave & 1, wn = wave >> 1;
    int m0 = blockIdx.x * 64;

    v8f acc[2][4];
#pragma unroll
    for (int i = 0; i < 2; ++i)
#pragma unroll
        for (int j = 0; j < 4; ++j)
#pragma unroll
            for (int k = 0; k < 8; ++k) acc[i][j][k] = 0.f;

    // Per-thread 16B chunk geometry: A = 256 chunks (1/thread), B = 1024 (4/thread)
    int arow = tid >> 2, acol = (tid & 3) * 8;
    int brow[4], bcol[4];
    unsigned ldsA[2], ldsB[2][4];
#pragma unroll
    for (int i = 0; i < 4; ++i) {
        int c = tid + i * 256;
        brow[i] = c >> 2;
        bcol[i] = (c & 3) * 8;
    }
#pragma unroll
    for (int buf = 0; buf < 2; ++buf) {
        ldsA[buf] = ldsAddr(&As[buf][arow][acol]);
#pragma unroll
        for (int i = 0; i < 4; ++i) ldsB[buf][i] = ldsAddr(&Bs[buf][brow[i]][bcol[i]]);
    }

    auto stage = [&](int kt, int buf) {
        asyncCopy16(ldsA[buf], Xh + (size_t)(m0 + arow) * DIMX + kt * 32 + acol);
#pragma unroll
        for (int i = 0; i < 4; ++i)
            asyncCopy16(ldsB[buf][i], Ph + (size_t)brow[i] * DIMX + kt * 32 + bcol[i]);
    };

    stage(0, 0);
    waitAsync0();
    __syncthreads();
    for (int kt = 0; kt < 8; ++kt) {
        int buf = kt & 1;
        if (kt + 1 < 8) stage(kt + 1, buf ^ 1);
        v16h af[2], bf[4];
#pragma unroll
        for (int i = 0; i < 2; ++i) af[i] = ldsFrag(&As[buf][wm * 32 + i * 16][0], 32, lane);
#pragma unroll
        for (int j = 0; j < 4; ++j) bf[j] = ldsFrag(&Bs[buf][wn * 64 + j * 16][0], 32, lane);
#pragma unroll
        for (int i = 0; i < 2; ++i)
#pragma unroll
            for (int j = 0; j < 4; ++j) acc[i][j] = wmma16(af[i], bf[j], acc[i][j]);
        if (kt + 1 < 8) waitAsync0();
        __syncthreads();
    }

    // Epilogue: complex exp -> 4 real fp16 features per accumulated dot product.
#pragma unroll
    for (int j = 0; j < 4; ++j) {
        int r = wn * 64 + j * 16 + (lane & 15);
        const float* c = rfc + r * 8;
        float cdr = c[0], cdi = c[1], hr = c[2], hi = c[3], mg2 = c[4];
#pragma unroll
        for (int i = 0; i < 2; ++i) {
#pragma unroll
            for (int g = 0; g < 8; ++g) {
                int m = m0 + wm * 32 + i * 16 + ((lane >> 4) << 3) + g;
                float P  = acc[i][j][g];
                float sq = sqx[m];
                float are =  cdr * P - sq * hr - mg2;
                float aim =  cdi * P - sq * hi;
                float bre = -cdr * P - sq * hr - mg2;
                float bim = -cdi * P - sq * hi;
                float ep = SFEAT * __expf(are);
                float em = SFEAT * __expf(bre);
                float sp, cp, sm, cm;
                __sincosf(aim, &sp, &cp);
                __sincosf(bim, &sm, &cm);
                _Float16* fr = F + (size_t)m * KBIG;
                fr[r]       = (_Float16)(ep * cp);
                fr[256 + r] = (_Float16)(em * cm);
                fr[512 + r] = (_Float16)(ep * sp);
                fr[768 + r] = (_Float16)(em * sm);
            }
        }
    }
}

// ---------------------------------------------------------------------------
// Kernel 5: out = F (N x 1024, f16) @ W^T (1024 x 1024, f16), f32 accumulate.
//   Block tile 128x128, 8 waves as 4(M) x 2(N); wave tile 32x64 (8 WMMA accs).
//   K = 1024 in 32-wide steps; async global->LDS double buffering.
// ---------------------------------------------------------------------------
__global__ __launch_bounds__(256) void gemm_kernel(const _Float16* __restrict__ F,
                                                   const _Float16* __restrict__ W,
                                                   float* __restrict__ out) {
    __shared__ __align__(16) _Float16 As[2][128][32];   // 16 KB
    __shared__ __align__(16) _Float16 Bs[2][128][32];   // 16 KB
    int tid = threadIdx.x, lane = tid & 31, wave = tid >> 5;
    int wm = wave & 3, wn = wave >> 2;
    int m0 = blockIdx.x * 128, n0 = blockIdx.y * 128;

    v8f acc[2][4];
#pragma unroll
    for (int i = 0; i < 2; ++i)
#pragma unroll
        for (int j = 0; j < 4; ++j)
#pragma unroll
            for (int k = 0; k < 8; ++k) acc[i][j][k] = 0.f;

    // Per-thread 16B chunk geometry: A and B each 512 chunks (2/thread)
    int r0 = tid >> 2,          c0 = (tid & 3) * 8;
    int r1 = (tid + 256) >> 2,  c1 = ((tid + 256) & 3) * 8;
    unsigned ldsA[2][2], ldsB[2][2];
#pragma unroll
    for (int buf = 0; buf < 2; ++buf) {
        ldsA[buf][0] = ldsAddr(&As[buf][r0][c0]);
        ldsA[buf][1] = ldsAddr(&As[buf][r1][c1]);
        ldsB[buf][0] = ldsAddr(&Bs[buf][r0][c0]);
        ldsB[buf][1] = ldsAddr(&Bs[buf][r1][c1]);
    }

    auto stage = [&](int kt, int buf) {
        asyncCopy16(ldsA[buf][0], F + (size_t)(m0 + r0) * KBIG + kt * 32 + c0);
        asyncCopy16(ldsA[buf][1], F + (size_t)(m0 + r1) * KBIG + kt * 32 + c1);
        asyncCopy16(ldsB[buf][0], W + (size_t)(n0 + r0) * KBIG + kt * 32 + c0);
        asyncCopy16(ldsB[buf][1], W + (size_t)(n0 + r1) * KBIG + kt * 32 + c1);
    };

    stage(0, 0);
    waitAsync0();
    __syncthreads();
    for (int kt = 0; kt < 32; ++kt) {
        int buf = kt & 1;
        if (kt + 1 < 32) stage(kt + 1, buf ^ 1);
        if (kt + 2 < 32) {  // L2 prefetch of the tile after next (global_prefetch_b8)
            __builtin_prefetch(F + (size_t)(m0 + (tid >> 1)) * KBIG + (kt + 2) * 32, 0, 1);
            __builtin_prefetch(W + (size_t)(n0 + (tid >> 1)) * KBIG + (kt + 2) * 32, 0, 1);
        }
        v16h af[2], bf[4];
#pragma unroll
        for (int i = 0; i < 2; ++i) af[i] = ldsFrag(&As[buf][wm * 32 + i * 16][0], 32, lane);
#pragma unroll
        for (int j = 0; j < 4; ++j) bf[j] = ldsFrag(&Bs[buf][wn * 64 + j * 16][0], 32, lane);
#pragma unroll
        for (int i = 0; i < 2; ++i)
#pragma unroll
            for (int j = 0; j < 4; ++j) acc[i][j] = wmma16(af[i], bf[j], acc[i][j]);
        if (kt + 1 < 32) waitAsync0();
        __syncthreads();
    }

#pragma unroll
    for (int i = 0; i < 2; ++i) {
#pragma unroll
        for (int j = 0; j < 4; ++j) {
            int n = n0 + wn * 64 + j * 16 + (lane & 15);
#pragma unroll
            for (int g = 0; g < 8; ++g) {
                int m = m0 + wm * 32 + i * 16 + ((lane >> 4) << 3) + g;
                out[(size_t)m * OUTD + n] = acc[i][j][g];
            }
        }
    }
}

// ---------------------------------------------------------------------------
// Launch.  Workspace layout (bytes), total ~192 MB:
//   RFC 0..8K | SQX 16K..272K | PH 512K..640K | XH 1M..33M | WB 34M..36M | F 64M..192M
// ---------------------------------------------------------------------------
extern "C" void kernel_launch(void* const* d_in, const int* in_sizes, int n_in,
                              void* d_out, int out_size, void* d_ws, size_t ws_size,
                              hipStream_t stream) {
    (void)in_sizes; (void)n_in; (void)out_size; (void)ws_size;
    const float* x    = (const float*)d_in[0];
    const float* iw   = (const float*)d_in[1];
    const float* bvec = (const float*)d_in[2];
    const float* proj = (const float*)d_in[3];
    const float* xr   = (const float*)d_in[4];
    const float* xi   = (const float*)d_in[5];
    float* out = (float*)d_out;

    char* ws = (char*)d_ws;
    float*    rfc = (float*)   (ws + 0);
    float*    sqx = (float*)   (ws + 16384);
    _Float16* Ph  = (_Float16*)(ws + 524288);
    _Float16* Xh  = (_Float16*)(ws + (1u << 20));
    _Float16* WB  = (_Float16*)(ws + 35651584u);
    _Float16* F   = (_Float16*)(ws + (1u << 26));

    rf_kernel   <<<256,              256, 0, stream>>>(proj, xr, xi, rfc, Ph);
    xprep_kernel<<<NROW / 8,         256, 0, stream>>>(x, Xh, sqx);
    wfeat_kernel<<<OUTD,             256, 0, stream>>>(iw, bvec, proj, rfc, WB);
    xfeat_kernel<<<NROW / 64,        256, 0, stream>>>(Xh, Ph, rfc, sqx, F);
    gemm_kernel <<<dim3(NROW / 128, OUTD / 128), 256, 0, stream>>>(F, WB, out);
}
